// EntAwareSelfAttnEncoder_13340168421489
// MI455X (gfx1250) — matrix operand order repaired and checked by reading, not compiled, not run on hardware
//
#include <hip/hip_runtime.h>
#include <hip/hip_bf16.h>

typedef __attribute__((ext_vector_type(16))) _Float16 v16h;
typedef __attribute__((ext_vector_type(8)))  float    v8f;

#define Bb 8
#define Ll 256
#define Hh 128

// Branch-free tanh: hardware v_tanh_f32 if available, else
// 1 - 2*rcp(exp(2x)+1)  (exact limits at +-inf, no EXEC divergence).
__device__ __forceinline__ float fast_tanh(float x) {
#if __has_builtin(__builtin_amdgcn_tanhf)
    return __builtin_amdgcn_tanhf(x);
#else
    float e = __expf(2.0f * x);                    // v_mul + v_exp_f32
    float r = __builtin_amdgcn_rcpf(e + 1.0f);     // v_rcp_f32
    return __builtin_fmaf(-2.0f, r, 1.0f);
#endif
}

// ---------------- workspace layout (bytes) ----------------
// agg  : f32 [B*H]        @ 0          (4 KB)
// term : f32 [B*L*H]      @ 4096       (1 MB)
// docp : f16 [B*L*H]      @ 1052672    (512 KB)  (A-fragment permuted)
// Wp   : f16 [8*4*32*16]  @ 1576960    (32 KB)   (B-fragment packed W1a)

// agg[b,h] = sum_l mask[b,l] * wei[b,l,h]
__global__ void k_agg(const float* __restrict__ wei,
                      const float* __restrict__ mask,
                      float* __restrict__ agg) {
    int b = blockIdx.x;
    int h = threadIdx.x;
    const float* wp = wei + (size_t)b * Ll * Hh + h;
    const float* mp = mask + b * Ll;
    float acc = 0.f;
    for (int l = 0; l < Ll; ++l) acc += mp[l] * wp[l * Hh];
    agg[b * Hh + h] = acc;
}

// docp[row, 32c+16hi+t] = f16(doc[row, 32c + 8hi + (t<8 ? t : 8+t)])
// Matches the ISA "16-bit A-Matrix 16x32" per-lane K ordering.
__global__ void k_docp(const float* __restrict__ doc,
                       _Float16* __restrict__ docp) {
    int idx = blockIdx.x * 256 + threadIdx.x;       // 0 .. B*L*H-1
    int p   = idx & (Hh - 1);
    int row = idx >> 7;
    int c   = p >> 5;
    int rem = p & 31;
    int hi  = (rem >> 4) & 1;
    int t   = rem & 15;
    int srch = 32 * c + 8 * hi + (t < 8 ? t : 8 + t);
    docp[idx] = (_Float16)doc[row * Hh + srch];
}

// Wp[((n*4+c)*32 + lane)*16 + t] = f16(W1a[32c + 8hi + (t<8 ? t : 8+t), 16n + lo])
// Dense-B striping: same per-lane K ordering as the A fragment (docp).
__global__ void k_packw(const float* __restrict__ W1,
                        _Float16* __restrict__ Wp) {
    int idx  = blockIdx.x * 256 + threadIdx.x;      // 0 .. 16383
    int t    = idx & 15;
    int lane = (idx >> 4) & 31;
    int c    = (idx >> 9) & 3;
    int n    = idx >> 11;
    int hi   = lane >> 4;
    int lo   = lane & 15;
    int h    = 32 * c + 8 * hi + (t < 8 ? t : 8 + t);
    int k    = 16 * n + lo;
    Wp[idx]  = (_Float16)W1[h * Hh + k];            // W1a = W1[:128]
}

// term[row,k] = b1[k] + sum_h (doc[row,h]*agg[b,h]) * W1b[h,k]
__global__ void k_term(const float* __restrict__ doc,
                       const float* __restrict__ agg,
                       const float* __restrict__ W1,
                       const float* __restrict__ b1,
                       float* __restrict__ term) {
    __shared__ float s[Hh];
    int row = blockIdx.x;            // b*L + i
    int b   = row >> 8;
    int k   = threadIdx.x;
    s[k] = doc[row * Hh + k] * agg[b * Hh + k];
    __syncthreads();
    float acc = b1[k];
    const float* w = W1 + Hh * Hh;   // W1b = W1[128:]
    for (int h = 0; h < Hh; ++h) acc += s[h] * w[h * Hh + k];
    term[row * Hh + k] = acc;
}

// One block per (b,i). 8 waves; each wave owns j-tiles {wave, wave+8}.
__global__ void __launch_bounds__(256)
k_main(const _Float16* __restrict__ docp,
       const _Float16* __restrict__ Wp,
       const float* __restrict__ term,
       const float* __restrict__ doc,
       const float* __restrict__ dmask,
       const float* __restrict__ W2,
       const float* __restrict__ b2p,
       float* __restrict__ out) {
    __shared__ float s_score[Ll];
    __shared__ float s_w[Ll];
    __shared__ float s_red[8];

    const int tid  = threadIdx.x;
    const int lane = tid & 31;
    const int wave = tid >> 5;
    const int lo   = lane & 15;
    const int hi   = lane >> 4;
    const int bi   = blockIdx.x;     // b*L + i
    const int b    = bi >> 8;
    const float dmi = dmask[bi];
    const float b2  = b2p[0];

    float w2r[8], tr[8];
    #pragma unroll
    for (int n = 0; n < 8; ++n) {
        w2r[n] = W2[16 * n + lo];
        tr[n]  = term[bi * Hh + 16 * n + lo];
    }

    v16h di[4];
    #pragma unroll
    for (int c = 0; c < 4; ++c)
        di[c] = *(const v16h*)(docp + (size_t)bi * Hh + 32 * c + 16 * hi);

    #pragma unroll 1
    for (int jj = 0; jj < 2; ++jj) {
        const int jt = wave + 8 * jj;
        const int j0 = jt * 16;

        // A fragment: a[m=j_local, k=h] = doc_i[h] * doc_j[h]  (f16)
        v16h a[4];
        #pragma unroll
        for (int c = 0; c < 4; ++c) {
            v16h dj = *(const v16h*)(docp + (size_t)(b * Ll + j0 + lo) * Hh
                                           + 32 * c + 16 * hi);
            a[c] = di[c] * dj;                      // v_pk_mul_f16 x8
        }

        float accv[8];
        #pragma unroll
        for (int r = 0; r < 8; ++r) accv[r] = 0.f;

        // keep as a real loop: only one k-tile's B fragments (32 VGPRs) live
        #pragma unroll 1
        for (int n = 0; n < 8; ++n) {
            v8f cc;
            #pragma unroll
            for (int r = 0; r < 8; ++r) cc[r] = tr[n];   // broadcast term over j
            #pragma unroll
            for (int c = 0; c < 4; ++c) {
                v16h bf = *(const v16h*)(Wp + (size_t)((n * 4 + c) * 32 + lane) * 16);
                cc = __builtin_amdgcn_wmma_f32_16x16x32_f16(
                        false, a[c], false, bf, (short)0, cc, false, false);
            }
            #pragma unroll
            for (int r = 0; r < 8; ++r) accv[r] += fast_tanh(cc[r]) * w2r[n];
        }

        // reduce over the 16 k-lanes (lo); masks 1..8 stay within half-wave
        #pragma unroll
        for (int m = 1; m < 16; m <<= 1) {
            #pragma unroll
            for (int r = 0; r < 8; ++r)
                accv[r] += __shfl_xor(accv[r], m, 32);
        }
        if (lo == 0) {
            #pragma unroll
            for (int r = 0; r < 8; ++r) {
                int j = j0 + r + 8 * hi;            // C layout: M = r + 8*hi
                float sc  = (accv[r] + b2) * 0.08838834764831845f; // 1/sqrt(128)
                float msk = dmi * dmask[b * Ll + j];
                s_score[j] = (msk == 0.f) ? -1.0e9f : sc;
            }
        }
    }
    __syncthreads();

    // ---- softmax over j (256) ----
    const int j = tid;
    float sc = s_score[j];
    float v = sc;
    #pragma unroll
    for (int m = 16; m >= 1; m >>= 1) v = fmaxf(v, __shfl_xor(v, m, 32));
    if (lane == 0) s_red[wave] = v;
    __syncthreads();
    if (tid == 0) {
        float mx = s_red[0];
        for (int q = 1; q < 8; ++q) mx = fmaxf(mx, s_red[q]);
        s_red[0] = mx;
    }
    __syncthreads();
    const float mx = s_red[0];
    const float e  = __expf(sc - mx);
    float sv = e;
    #pragma unroll
    for (int m = 16; m >= 1; m >>= 1) sv += __shfl_xor(sv, m, 32);
    __syncthreads();
    if (lane == 0) s_red[wave] = sv;
    __syncthreads();
    if (tid == 0) {
        float su = 0.f;
        for (int q = 0; q < 8; ++q) su += s_red[q];
        s_red[0] = su;
    }
    __syncthreads();
    const float rsu = __builtin_amdgcn_rcpf(s_red[0]);
    const float mj  = dmi * dmask[b * Ll + j];
    s_w[j] = e * rsu * mj;
    __syncthreads();

    // ---- out[b,i,h] = sum_j w[j] * doc[b,j,h] ----
    if (tid < Hh) {
        const int h = tid;
        float acc = 0.f;
        for (int jq = 0; jq < Ll; ++jq)
            acc += s_w[jq] * doc[(size_t)(b * Ll + jq) * Hh + h];
        out[(size_t)bi * Hh + h] = acc;
    }
}

extern "C" void kernel_launch(void* const* d_in, const int* in_sizes, int n_in,
                              void* d_out, int out_size, void* d_ws, size_t ws_size,
                              hipStream_t stream) {
    (void)in_sizes; (void)n_in; (void)out_size; (void)ws_size;
    const float* wei   = (const float*)d_in[0];
    const float* wmask = (const float*)d_in[1];
    const float* doc   = (const float*)d_in[2];
    const float* dmsk  = (const float*)d_in[3];
    const float* W1    = (const float*)d_in[4];
    const float* b1    = (const float*)d_in[5];
    const float* W2    = (const float*)d_in[6];
    const float* b2    = (const float*)d_in[7];
    float* out = (float*)d_out;

    char* ws = (char*)d_ws;
    float*    agg  = (float*)(ws + 0);
    float*    term = (float*)(ws + 4096);
    _Float16* docp = (_Float16*)(ws + 4096 + 1048576);
    _Float16* Wp   = (_Float16*)(ws + 4096 + 1048576 + 524288);

    k_agg  <<<Bb, Hh, 0, stream>>>(wei, wmask, agg);
    k_docp <<<(Bb * Ll * Hh) / 256, 256, 0, stream>>>(doc, docp);
    k_packw<<<64, 256, 0, stream>>>(W1, Wp);
    k_term <<<Bb * Ll, Hh, 0, stream>>>(doc, agg, W1, b1, term);
    k_main <<<Bb * Ll, 256, 0, stream>>>(docp, Wp, term, doc, dmsk, W2, b2, out);
}
